// GRASPLayer_16681652977859
// MI455X (gfx1250) — matrix operand order repaired and checked by reading, not compiled, not run on hardware
//
#include <hip/hip_runtime.h>
#include <hip/hip_bf16.h>
#include <math.h>

#define DEVFN __device__ __forceinline__

typedef __bf16 bf16_t;
typedef __attribute__((ext_vector_type(16))) __bf16 v16bf;
typedef __attribute__((ext_vector_type(8)))  __bf16 v8bf;
typedef __attribute__((ext_vector_type(8)))  float  v8f;

constexpr int Bsz = 8192;
constexpr int T   = 128;
constexpr int IN  = 64;
constexpr int H   = 256;
constexpr int NC  = 12;          // K-means clusters
constexpr int KM_ITERS = 25;
constexpr int KTOT = IN + H;     // 320 : concat [x_t | h] along K
constexpr int NCOL = 1024;       // r(256) | z(256) | xn(256) | hn(256)
constexpr int KT   = KTOT / 32;  // 10 k-tiles of 32
constexpr int NT   = NCOL / 16;  // 64 n-tiles of 16

DEVFN bf16_t f2bf(float f) {
  union { float f; unsigned u; } v; v.f = f;
  unsigned r = v.u + 0x7FFFu + ((v.u >> 16) & 1u);   // round-to-nearest-even
  union { unsigned short s; bf16_t b; } o;
  o.s = (unsigned short)(r >> 16);
  return o.b;
}
DEVFN float sigmoidf_(float x) { return 1.0f / (1.0f + __expf(-x)); }
DEVFN unsigned wang(unsigned s) { s=(s^61u)^(s>>16); s*=9u; s^=s>>4; s*=0x27d4eb2du; s^=s>>15; return s; }

// ---------------------------------------------------------------------------
// Pack GRU weights (fp32 -> bf16) into WMMA B-fragment order.
// Logical W[k][c], k in [0,320), c in [0,1024):
//   c in [0,512)    : fused r/z columns : k<64 ? wih[c][k] : whh[c][k-64]
//   c in [512,768)  : xn columns       : k<64 ? wih[512+cc][k] : 0
//   c in [768,1024) : hn columns       : k<64 ? 0 : whh[512+cc][k-64]
// Fragment (kt,nt) holds 32x16; lane = (kk/16)*16 + n, elem = kk%16 so each
// lane's 16 bf16 are one aligned 32B load. [kt][nt] order => a wave's 8
// fragment loads per k-tile are sequential in memory.
// ---------------------------------------------------------------------------
__global__ void pack_weights(const float* __restrict__ wih,
                             const float* __restrict__ whh,
                             bf16_t* __restrict__ Wf) {
  int idx = blockIdx.x * blockDim.x + threadIdx.x;
  if (idx >= KTOT * NCOL) return;
  int k = idx / NCOL;
  int c = idx - k * NCOL;
  float val;
  if (c < 512) {
    val = (k < IN) ? wih[c * IN + k] : whh[c * H + (k - IN)];
  } else if (c < 768) {
    int row = 512 + (c - 512);
    val = (k < IN) ? wih[row * IN + k] : 0.0f;
  } else {
    int row = 512 + (c - 768);
    val = (k < IN) ? 0.0f : whh[row * H + (k - IN)];
  }
  int kt = k >> 5, kk = k & 31;
  int nt = c >> 4, n  = c & 15;
  int lane = ((kk >> 4) << 4) | n;
  int elem = kk & 15;
  Wf[(size_t)((kt * NT + nt) * 32 + lane) * 16 + elem] = f2bf(val);
}

// convert x (fp32) to bf16 once: halves the dominant HBM read stream
__global__ void convert_x(const float* __restrict__ x, bf16_t* __restrict__ xbf,
                          int n4) {
  int i = blockIdx.x * blockDim.x + threadIdx.x;
  if (i >= n4) return;
  const float4 v = ((const float4*)x)[i];
  union { bf16_t b[4]; unsigned long long u; } o;
  o.b[0] = f2bf(v.x); o.b[1] = f2bf(v.y); o.b[2] = f2bf(v.z); o.b[3] = f2bf(v.w);
  ((unsigned long long*)xbf)[i] = o.u;
}

__global__ void zero_f32(float* p, int n) {
  int i = blockIdx.x * blockDim.x + threadIdx.x;
  if (i < n) p[i] = 0.0f;
}

// ---------------------------------------------------------------------------
// One GRU time step, fused input projection + recurrence + gates.
// grid = B/16 blocks, 256 threads (8 waves). Wave w owns n-tiles [8w, 8w+8).
// A tile [16 x 320] bf16 staged straight from global bf16 buffers into LDS
// with GLOBAL_LOAD_ASYNC_TO_LDS_B128 (ASYNCcnt), no VGPR bounce, no convert.
// ---------------------------------------------------------------------------
__global__ __launch_bounds__(256)
void gru_step(const bf16_t* __restrict__ xbf, const bf16_t* __restrict__ Wf,
              const float* __restrict__ bih, const float* __restrict__ bhh,
              const float* __restrict__ h_in, const bf16_t* __restrict__ hbf_in,
              float* __restrict__ h_out, bf16_t* __restrict__ hbf_out,
              int step) {
  // A-stage (16x320 bf16 = 10 KB) and gate buffer (16x1024 f32 = 64 KB)
  // live in the same LDS (disjoint in time).
  __shared__ __align__(16) unsigned char smem[16 * NCOL * 4];
  bf16_t* ldsA = (bf16_t*)smem;
  float*  ldsG = (float*)smem;

  const int tid  = threadIdx.x;
  const int lane = tid & 31;
  const int wave = tid >> 5;
  const int rowBase = blockIdx.x * 16;

  // pull this wave's weight stream toward L2 (global_prefetch_b8)
  __builtin_prefetch(Wf + (size_t)(wave * 8) * 512, 0, 1);

  // ---- async A-tile staging: 16 rows x 640B = 640 16-byte chunks ----------
  // row r: chunks 0..7  <- xbf[(row*T + step)*64 ..]   (128 B)
  //        chunks 8..39 <- hbf_in[row*256 ..]          (512 B)
  unsigned ldsBase = (unsigned)(uintptr_t)(&ldsA[0]);   // low 32 bits = LDS offset
  for (int q = tid; q < 16 * 40; q += 256) {
    int r = q / 40, c = q - r * 40;
    int gr = rowBase + r;
    const bf16_t* src = (c < 8)
        ? (xbf + ((size_t)gr * T + step) * IN + c * 8)
        : (hbf_in + (size_t)gr * H + (c - 8) * 8);
    unsigned lofs = ldsBase + (unsigned)(q * 16);
    unsigned long long ga = (unsigned long long)(uintptr_t)src;
    asm volatile("global_load_async_to_lds_b128 %0, %1, off"
                 :: "v"(lofs), "v"(ga) : "memory");
  }
  asm volatile("s_wait_asynccnt 0" ::: "memory");
  __syncthreads();

  const int half = lane >> 4;   // lane group (K split per 16-bit A layout)
  const int arow = lane & 15;   // M row for this lane

  v8f acc[8];
  #pragma unroll
  for (int j = 0; j < 8; ++j) {
    #pragma unroll
    for (int e = 0; e < 8; ++e) acc[j][e] = 0.0f;
  }

  #pragma unroll
  for (int kt = 0; kt < KT; ++kt) {
    // preload all 8 B fragments for this k-tile (one clause, one wait),
    // then fire 8 independent WMMAs back-to-back.
    v16bf b[8];
    #pragma unroll
    for (int j = 0; j < 8; ++j) {
      int nt = wave * 8 + j;
      b[j] = *(const v16bf*)(Wf + (size_t)((kt * NT + nt) * 32 + lane) * 16);
    }
    // A fragment: elems 0..15 -> K = half*16 + 0..15
    const bf16_t* ap = &ldsA[arow * KTOT + (kt << 5) + (half << 4)];
    v8bf alo = *(const v8bf*)(ap);
    v8bf ahi = *(const v8bf*)(ap + 8);
    v16bf a = __builtin_shufflevector(alo, ahi, 0,1,2,3,4,5,6,7,8,9,10,11,12,13,14,15);
    #pragma unroll
    for (int j = 0; j < 8; ++j)
      acc[j] = __builtin_amdgcn_wmma_f32_16x16x32_bf16(
          false, a, false, b[j], (short)0, acc[j], false, false);
  }
  __syncthreads();   // everyone done reading ldsA before ldsG overwrites it

  // Scatter accumulators. C/D layout: VGPR v, lane L -> M = v+8*(L/16), N = L%16.
  #pragma unroll
  for (int j = 0; j < 8; ++j) {
    int col = (wave * 8 + j) * 16 + (lane & 15);
    #pragma unroll
    for (int v = 0; v < 8; ++v)
      ldsG[(v + (half << 3)) * NCOL + col] = acc[j][v];
  }
  __syncthreads();

  // Gate math: r = sig(xr+hr+b), z = sig(xz+hz+b), n = tanh(xn + r*hn)
  for (int t = tid; t < 16 * H; t += 256) {
    int r = t >> 8, c = t & 255;
    float gr_ = ldsG[r * NCOL + c]       + bih[c]       + bhh[c];
    float gz  = ldsG[r * NCOL + 256 + c] + bih[256 + c] + bhh[256 + c];
    float xn  = ldsG[r * NCOL + 512 + c] + bih[512 + c];
    float hn  = ldsG[r * NCOL + 768 + c] + bhh[512 + c];
    float rg = sigmoidf_(gr_);
    float zg = sigmoidf_(gz);
    float ng = tanhf(xn + rg * hn);
    float hp = h_in[(size_t)(rowBase + r) * H + c];
    float ho = (1.0f - zg) * ng + zg * hp;
    h_out[(size_t)(rowBase + r) * H + c] = ho;
    hbf_out[(size_t)(rowBase + r) * H + c] = f2bf(ho);   // bf16 copy for next step
  }
}

// ---------------------------------------------------------------------------
// K-means / GCN / blend (bandwidth noise vs the GRU)
// ---------------------------------------------------------------------------
__global__ void init_centers(const float* __restrict__ h, float* __restrict__ cen) {
  int i = blockIdx.x * blockDim.x + threadIdx.x;
  if (i >= NC * H) return;
  int k = i / H, c = i - k * H;
  int row = (k * 683 + 37) & (Bsz - 1);   // deterministic distinct seed rows
  cen[i] = h[(size_t)row * H + c];
}

__global__ __launch_bounds__(256)
void kmeans_assign(const float* __restrict__ h, const float* __restrict__ centers,
                   int* __restrict__ codes) {
  __shared__ float cen[NC * H];
  for (int t = threadIdx.x; t < NC * H; t += 256) cen[t] = centers[t];
  __syncthreads();
  int lane = threadIdx.x & 31, wave = threadIdx.x >> 5;
  int row = blockIdx.x * 8 + wave;
  float hv[8];
  #pragma unroll
  for (int i = 0; i < 8; ++i) hv[i] = h[(size_t)row * H + lane + 32 * i];
  int best = 0; float bestd = 3.4e38f;
  for (int k = 0; k < NC; ++k) {
    float d = 0.0f;
    #pragma unroll
    for (int i = 0; i < 8; ++i) {
      float t = hv[i] - cen[k * H + lane + 32 * i];
      d += t * t;
    }
    #pragma unroll
    for (int s = 16; s > 0; s >>= 1) d += __shfl_xor(d, s, 32);
    if (d < bestd) { bestd = d; best = k; }
  }
  if (lane == 0) codes[row] = best;
}

__global__ __launch_bounds__(256)
void kmeans_accum(const float* __restrict__ h, const int* __restrict__ codes,
                  float* __restrict__ csum, float* __restrict__ ccnt) {
  int lane = threadIdx.x & 31, wave = threadIdx.x >> 5;
  int row = blockIdx.x * 8 + wave;
  int k = codes[row];
  #pragma unroll
  for (int i = 0; i < 8; ++i)
    atomicAdd(&csum[k * H + lane + 32 * i], h[(size_t)row * H + lane + 32 * i]);
  if (lane == 0) atomicAdd(&ccnt[k], 1.0f);
}

__global__ void kmeans_divide(const float* __restrict__ csum,
                              const float* __restrict__ ccnt,
                              float* __restrict__ centers) {
  int i = blockIdx.x * blockDim.x + threadIdx.x;
  if (i >= NC * H) return;
  float c = ccnt[i / H];
  centers[i] = csum[i] / (c > 0.5f ? c : 1.0f);
}

__global__ __launch_bounds__(256)
void gcn(const float* __restrict__ centers,
         const float* __restrict__ w1, const float* __restrict__ b1,
         const float* __restrict__ w2, const float* __restrict__ b2,
         float* __restrict__ h2out) {
  __shared__ float sc[NC * H];
  __shared__ float sh1[NC * H];
  for (int t = threadIdx.x; t < NC * H; t += 256) sc[t] = centers[t];
  __syncthreads();
  int c = threadIdx.x;
  for (int k = 0; k < NC; ++k) {
    float acc = b1[c];
    for (int j = 0; j < H; ++j) acc += sc[k * H + j] * w1[j * H + c];
    sh1[k * H + c] = fmaxf(acc, 0.0f);
  }
  __syncthreads();
  for (int k = 0; k < NC; ++k) {
    float acc = b2[c];
    for (int j = 0; j < H; ++j) acc += sh1[k * H + j] * w2[j * H + c];
    h2out[k * H + c] = fmaxf(acc, 0.0f);
  }
}

// straight-through gumbel-softmax hard scores == one_hot(argmax(logits + g))
__global__ __launch_bounds__(256)
void score_argmax(const float* __restrict__ h, const float* __restrict__ centers,
                  int* __restrict__ amax) {
  __shared__ float cen[NC * H];
  for (int t = threadIdx.x; t < NC * H; t += 256) cen[t] = centers[t];
  __syncthreads();
  int lane = threadIdx.x & 31, wave = threadIdx.x >> 5;
  int row = blockIdx.x * 8 + wave;
  float hv[8];
  #pragma unroll
  for (int i = 0; i < 8; ++i) hv[i] = h[(size_t)row * H + lane + 32 * i];
  int best = 0; float bestv = -3.4e38f;
  for (int k = 0; k < NC; ++k) {
    float d = 0.0f;
    #pragma unroll
    for (int i = 0; i < 8; ++i) d += hv[i] * cen[k * H + lane + 32 * i];
    #pragma unroll
    for (int s = 16; s > 0; s >>= 1) d += __shfl_xor(d, s, 32);
    float e = fmaxf(d, 0.0f);
    float u = (float)(wang((unsigned)(row * NC + k) * 2654435761u) + 1u) * 2.3283064e-10f;
    float g = -__logf(-__logf(u + 1e-20f) + 1e-20f);
    float v = e + g;
    if (v > bestv) { bestv = v; best = k; }
  }
  if (lane == 0) amax[row] = best;
}

__global__ __launch_bounds__(256)
void final_blend(const float* __restrict__ h, const float* __restrict__ h2,
                 const int* __restrict__ amax,
                 const float* __restrict__ w1w, const float* __restrict__ w1b,
                 const float* __restrict__ w2w, const float* __restrict__ w2b,
                 float* __restrict__ out) {
  int lane = threadIdx.x & 31, wave = threadIdx.x >> 5;
  int row = blockIdx.x * 8 + wave;
  int k = amax[row];
  float d1 = 0.0f, d2 = 0.0f;
  float cl[8], hh[8];
  #pragma unroll
  for (int i = 0; i < 8; ++i) {
    cl[i] = h2[k * H + lane + 32 * i];
    hh[i] = h[(size_t)row * H + lane + 32 * i];
    d1 += cl[i] * w1w[lane + 32 * i];
    d2 += hh[i] * w2w[lane + 32 * i];
  }
  #pragma unroll
  for (int s = 16; s > 0; s >>= 1) {
    d1 += __shfl_xor(d1, s, 32);
    d2 += __shfl_xor(d2, s, 32);
  }
  float wa = sigmoidf_(d1 + w1b[0]);
  float wb = sigmoidf_(d2 + w2b[0]);
  float wf = wa / (wa + wb);
  #pragma unroll
  for (int i = 0; i < 8; ++i)
    out[(size_t)row * H + lane + 32 * i] = wf * cl[i] + (1.0f - wf) * hh[i];
}

// ---------------------------------------------------------------------------
extern "C" void kernel_launch(void* const* d_in, const int* in_sizes, int n_in,
                              void* d_out, int out_size, void* d_ws, size_t ws_size,
                              hipStream_t stream) {
  const float* x   = (const float*)d_in[0];
  const float* wih = (const float*)d_in[1];
  const float* whh = (const float*)d_in[2];
  const float* bih = (const float*)d_in[3];
  const float* bhh = (const float*)d_in[4];
  const float* g1w = (const float*)d_in[5];
  const float* g1b = (const float*)d_in[6];
  const float* g2w = (const float*)d_in[7];
  const float* g2b = (const float*)d_in[8];
  const float* w1w = (const float*)d_in[9];
  const float* w1b = (const float*)d_in[10];
  const float* w2w = (const float*)d_in[11];
  const float* w2b = (const float*)d_in[12];

  char* ws = (char*)d_ws;
  size_t off = 0;
  auto alloc = [&](size_t bytes) -> void* {
    void* p = ws + off;
    off = (off + bytes + 255) & ~(size_t)255;
    return p;
  };
  bf16_t* Wf    = (bf16_t*)alloc((size_t)KTOT * NCOL * 2);     // 640 KB packed weights
  bf16_t* xbf   = (bf16_t*)alloc((size_t)Bsz * T * IN * 2);    // 128 MB bf16 x
  float*  hA    = (float*)alloc((size_t)Bsz * H * 4);          // 8 MB hidden ping
  float*  hB    = (float*)alloc((size_t)Bsz * H * 4);          // 8 MB hidden pong
  bf16_t* hbfA  = (bf16_t*)alloc((size_t)Bsz * H * 2);         // 4 MB bf16 ping
  bf16_t* hbfB  = (bf16_t*)alloc((size_t)Bsz * H * 2);         // 4 MB bf16 pong
  float*  cen   = (float*)alloc((size_t)NC * H * 4);
  float*  csum  = (float*)alloc(((size_t)NC * H + 64) * 4);    // sums + counts
  float*  ccnt  = csum + NC * H;
  int*    codes = (int*)alloc((size_t)Bsz * 4);
  int*    amax  = (int*)alloc((size_t)Bsz * 4);
  float*  h2b   = (float*)alloc((size_t)NC * H * 4);

  // 1) pack weights, pre-convert x to bf16, zero h0 (fp32 and bf16 views)
  pack_weights<<<(KTOT * NCOL + 255) / 256, 256, 0, stream>>>(wih, whh, Wf);
  {
    int n4 = Bsz * T * IN / 4;
    convert_x<<<(n4 + 255) / 256, 256, 0, stream>>>(x, xbf, n4);
  }
  zero_f32<<<(Bsz * H + 255) / 256, 256, 0, stream>>>(hA, Bsz * H);
  zero_f32<<<(Bsz * H / 2 + 255) / 256, 256, 0, stream>>>((float*)hbfA, Bsz * H / 2);

  // 2) GRU recurrence: async bf16 A-staging + bf16 WMMA, ping-pong h
  for (int t = 0; t < T; ++t) {
    const float*  hin  = (t & 1) ? hB : hA;
    float*        hout = (t & 1) ? hA : hB;
    const bf16_t* bin  = (t & 1) ? hbfB : hbfA;
    bf16_t*       bout = (t & 1) ? hbfA : hbfB;
    gru_step<<<Bsz / 16, 256, 0, stream>>>(xbf, Wf, bih, bhh, hin, bin, hout, bout, t);
  }
  const float* hid = hA;   // T = 128 even -> final hidden lands in hA

  // 3) K-means: init; assign; 25x (update, assign); final update
  init_centers<<<(NC * H + 255) / 256, 256, 0, stream>>>(hid, cen);
  kmeans_assign<<<Bsz / 8, 256, 0, stream>>>(hid, cen, codes);
  for (int it = 0; it < KM_ITERS; ++it) {
    zero_f32<<<(NC * H + 64 + 255) / 256, 256, 0, stream>>>(csum, NC * H + NC);
    kmeans_accum<<<Bsz / 8, 256, 0, stream>>>(hid, codes, csum, ccnt);
    kmeans_divide<<<(NC * H + 255) / 256, 256, 0, stream>>>(csum, ccnt, cen);
    kmeans_assign<<<Bsz / 8, 256, 0, stream>>>(hid, cen, codes);
  }
  zero_f32<<<(NC * H + 64 + 255) / 256, 256, 0, stream>>>(csum, NC * H + NC);
  kmeans_accum<<<Bsz / 8, 256, 0, stream>>>(hid, codes, csum, ccnt);
  kmeans_divide<<<(NC * H + 255) / 256, 256, 0, stream>>>(csum, ccnt, cen);

  // 4) GCN on centers; hard gumbel argmax; gather + gated blend
  gcn<<<1, 256, 0, stream>>>(cen, g1w, g1b, g2w, g2b, h2b);
  score_argmax<<<Bsz / 8, 256, 0, stream>>>(hid, cen, amax);
  final_blend<<<Bsz / 8, 256, 0, stream>>>(hid, h2b, amax, w1w, w1b, w2w, w2b,
                                           (float*)d_out);
}
